// HET_align2_69776038691149
// MI455X (gfx1250) — matrix operand order repaired and checked by reading, not compiled, not run on hardware
//
#include <hip/hip_runtime.h>
#include <hip/hip_bf16.h>
#include <math.h>

// ---------------- CDNA5 WMMA types ----------------
typedef __attribute__((ext_vector_type(16))) __bf16 v16bf;
typedef __attribute__((ext_vector_type(8)))  __bf16 v8bf;
typedef __attribute__((ext_vector_type(8)))  float  v8f;

// ---------------- problem constants ----------------
#define KG_E   50000
#define KG_R   500
#define E_DIM  300
#define NNZ    400000
#define NS     304      // padded column stride (19 tiles of 16)
#define KRP    50016LL  // relation-GEMM K padded to 32 (1563*32)
#define KAP    320      // entity-GEMM K padded to 32 (10*32)
#define ALPHA1 0.1f
#define ALPHA2 0.3f
#define LRA    0.2f

// =====================================================================
// rowsum -> 1/sum (or 0)
__global__ void k_rowsum_inv(const float* __restrict__ r, float* __restrict__ out, int K) {
  __shared__ float red[256];
  int row = blockIdx.x;
  const float* p = r + (long long)row * K;
  float s = 0.f;
  for (int k = threadIdx.x; k < K; k += 256) s += p[k];
  red[threadIdx.x] = s; __syncthreads();
  for (int off = 128; off > 0; off >>= 1) {
    if (threadIdx.x < off) red[threadIdx.x] += red[threadIdx.x + off];
    __syncthreads();
  }
  if (threadIdx.x == 0) { float t = red[0]; out[row] = (t == 0.f) ? 0.f : 1.f / t; }
}

// =====================================================================
// Build transposed bf16 B matrices for the relation GEMM:
// btX[n*KRP + k] = bf16(e[k*es + n] * beX[k]),  zero padded (k>=50000 or n>=300)
__global__ void k_make_bt(const float* __restrict__ e, int es,
                          const float* __restrict__ beL, const float* __restrict__ beR,
                          __bf16* __restrict__ btL, __bf16* __restrict__ btR) {
  __shared__ float tl[16][17], tr[16][17];
  int k0 = blockIdx.x * 16, n0 = blockIdx.y * 16;
  int tx = threadIdx.x, ty = threadIdx.y;
  int k = k0 + ty, n = n0 + tx;
  float v = 0.f, bl = 0.f, br = 0.f;
  if (k < KG_E) {
    bl = beL[k]; br = beR[k];
    if (n < E_DIM) v = e[(long long)k * es + n];
  }
  tl[ty][tx] = v * bl;
  tr[ty][tx] = v * br;
  __syncthreads();
  long long o = (long long)(n0 + ty) * KRP + (k0 + tx);
  btL[o] = (__bf16)tl[tx][ty];
  btR[o] = (__bf16)tr[tx][ty];
}

// =====================================================================
// Relation GEMM: out[r,n] = relu( (sum_k A[r,k]*B[k,n]) * rinv[r] ) * atten[n]
// A: r_head/r_tail f32 [500,50000] row-major (converted bf16 in-register)
// Bt: bf16 [NS][KRP] (transposed, zero-padded)
// Each wave: one 16-row M tile x TWO 16-col N tiles over full K.
// Rows >= 500 are computed on a clamped row pointer and discarded (branch-free loop).
__global__ void __launch_bounds__(128) k_gemm_r(
    const float* __restrict__ A, const __bf16* __restrict__ Bt,
    const float* __restrict__ rinv, const float* __restrict__ atten,
    float* __restrict__ out) {
  int wave = threadIdx.x >> 5, lane = threadIdx.x & 31;
  int ntp = blockIdx.y * 4 + wave;               // N tile-pair, 0..9
  if (ntp >= 10) return;                         // wave-uniform exit
  int nt0 = ntp * 2;
  int nt1 = nt0 + 1;
  int nt1c = (nt1 < NS / 16) ? nt1 : (NS / 16 - 1);  // clamp 19th tile (discarded)
  int m0 = blockIdx.x * 16;
  int lm = lane & 15, hi = lane >> 4;
  int abase = hi * 16, bbase = hi * 16;
  int m = m0 + lm;
  int mc = (m < KG_R) ? m : (KG_R - 1);          // clamp; rows >= 500 discarded
  const float*  arow = A  + (long long)mc * KG_E;
  const __bf16* b0p  = Bt + (long long)(nt0  * 16 + lm) * KRP + bbase;
  const __bf16* b1p  = Bt + (long long)(nt1c * 16 + lm) * KRP + bbase;
  v8f acc0 = {}, acc1 = {};
  int kk = 0;
  for (; kk + 32 <= KG_E; kk += 32) {            // 1562 branch-free steps
    v8f f0 = *(const v8f*)(arow + kk + abase);
    v8f f1 = *(const v8f*)(arow + kk + 8 + abase);
    v16bf a;
    #pragma unroll
    for (int i = 0; i < 8; ++i) { a[i] = (__bf16)f0[i]; a[8 + i] = (__bf16)f1[i]; }
    v16bf vb0 = *(const v16bf*)(b0p + kk);
    v16bf vb1 = *(const v16bf*)(b1p + kk);
    acc0 = __builtin_amdgcn_wmma_f32_16x16x32_bf16(false, a, false, vb0, (short)0, acc0, false, false);
    acc1 = __builtin_amdgcn_wmma_f32_16x16x32_bf16(false, a, false, vb1, (short)0, acc1, false, false);
  }
  { // tail: K 49984..49999 valid; 50000..50015 must be zero (Bt already zero-padded)
    v16bf a = {};
    #pragma unroll
    for (int i = 0; i < 8; ++i) {
      int kA = kk + abase + i;
      if (kA < KG_E) a[i] = (__bf16)arow[kA];
      int kB = kk + 8 + abase + i;
      if (kB < KG_E) a[8 + i] = (__bf16)arow[kB];
    }
    v16bf vb0 = *(const v16bf*)(b0p + kk);
    v16bf vb1 = *(const v16bf*)(b1p + kk);
    acc0 = __builtin_amdgcn_wmma_f32_16x16x32_bf16(false, a, false, vb0, (short)0, acc0, false, false);
    acc1 = __builtin_amdgcn_wmma_f32_16x16x32_bf16(false, a, false, vb1, (short)0, acc1, false, false);
  }
  int n0 = nt0 * 16 + lm;
  float at0 = (n0 < E_DIM) ? atten[n0] : 0.f;
  #pragma unroll
  for (int j = 0; j < 8; ++j) {
    int r = m0 + hi * 8 + j;
    if (r < KG_R) {
      float v = acc0[j] * rinv[r];
      v = v > 0.f ? v : 0.f;
      out[(long long)r * NS + n0] = v * at0;
    }
  }
  if (nt1 < NS / 16) {                           // wave-uniform
    int n1 = n0 + 16;
    float at1 = (n1 < E_DIM) ? atten[n1] : 0.f;
    #pragma unroll
    for (int j = 0; j < 8; ++j) {
      int r = m0 + hi * 8 + j;
      if (r < KG_R) {
        float v = acc1[j] * rinv[r];
        v = v > 0.f ? v : 0.f;
        out[(long long)r * NS + n1] = v * at1;
      }
    }
  }
}

// =====================================================================
// Fused attention: per triple t (one wave each):
//  score = e[row].aL[rel] + e[col].aR[rel]; w = exp(-leakyrelu(score))
//  rowsum[row] += w ; acc[row,:] += w * e[col,:]
__global__ void __launch_bounds__(256) k_att(
    const float* __restrict__ e, int es,
    const float* __restrict__ aL, const float* __restrict__ aR,
    const int* __restrict__ eidx, const int* __restrict__ erel,
    float* __restrict__ acc, float* __restrict__ rowsum, int ntr) {
  int wave = threadIdx.x >> 5, lane = threadIdx.x & 31;
  long long t = (long long)blockIdx.x * 8 + wave;
  if (t >= ntr) return;
  int row = eidx[t], col = eidx[(long long)ntr + t], rel = erel[t];
  const float* er = e + (long long)row * es;
  const float* ec = e + (long long)col * es;
  const float* al = aL + (long long)rel * NS;
  const float* ar = aR + (long long)rel * NS;
  float s = 0.f;
  for (int d = lane; d < E_DIM; d += 32) s += er[d] * al[d] + ec[d] * ar[d];
  for (int off = 16; off > 0; off >>= 1) s += __shfl_xor(s, off, 32);
  float lr = s > 0.f ? s : LRA * s;
  float w = expf(-lr);
  if (lane == 0) atomicAdd(&rowsum[row], w);
  float* ao = acc + (long long)row * NS;
  for (int d = lane; d < E_DIM; d += 32) atomicAdd(&ao[d], w * ec[d]);
}

// =====================================================================
// e_next = e0 + alpha * relu(acc * inv_or_zero(rowsum))   (stride NS, pads=0)
__global__ void k_att_combine(const float* __restrict__ e0, const float* __restrict__ acc,
                              const float* __restrict__ rowsum, float alpha,
                              float* __restrict__ out) {
  long long i = (long long)blockIdx.x * blockDim.x + threadIdx.x;
  if (i >= (long long)KG_E * NS) return;
  int r = (int)(i / NS), d = (int)(i % NS);
  float v = 0.f;
  if (d < E_DIM) {
    float rs = rowsum[r];
    float inv = (rs == 0.f) ? 0.f : 1.f / rs;
    float x = acc[i] * inv;
    x = x > 0.f ? x : 0.f;
    v = e0[(long long)r * E_DIM + d] + alpha * x;
  }
  out[i] = v;
}

// =====================================================================
// f32 [KG_E, NS] -> bf16 A matrix [KG_E, KAP] (K zero-padded)
__global__ void k_cvt_a(const float* __restrict__ x, __bf16* __restrict__ a) {
  long long i = (long long)blockIdx.x * blockDim.x + threadIdx.x;
  if (i >= (long long)KG_E * KAP) return;
  int r = (int)(i / KAP), d = (int)(i % KAP);
  a[i] = (d < E_DIM) ? (__bf16)x[(long long)r * NS + d] : (__bf16)0.f;
}

// W f32 [300,300] -> transposed bf16 [NS][KAP]
__global__ void k_cvt_wt(const float* __restrict__ W, __bf16* __restrict__ wt) {
  long long i = (long long)blockIdx.x * blockDim.x + threadIdx.x;
  if (i >= (long long)NS * KAP) return;
  int n = (int)(i / KAP), k = (int)(i % KAP);
  wt[i] = (n < E_DIM && k < E_DIM) ? (__bf16)W[(long long)k * E_DIM + n] : (__bf16)0.f;
}

// =====================================================================
// Entity GEMM: out[m,n] = sum_k A[m,k]*W[k,n]; A bf16 [KG_E,KAP], Wt bf16 [NS][KAP]
__global__ void __launch_bounds__(128) k_gemm2(
    const __bf16* __restrict__ A, const __bf16* __restrict__ Wt, float* __restrict__ out) {
  int wave = threadIdx.x >> 5, lane = threadIdx.x & 31;
  int nt = blockIdx.y * 4 + wave;
  if (nt >= NS / 16) return;
  int lm = lane & 15, hi = lane >> 4;
  int m = blockIdx.x * 16 + lm;
  const __bf16* arow = A  + (long long)m * KAP + hi * 16;
  const __bf16* bcol = Wt + (long long)(nt * 16 + lm) * KAP + hi * 16;
  v8f acc = {};
  #pragma unroll
  for (int kk = 0; kk < KAP; kk += 32) {
    v8bf a0 = *(const v8bf*)(arow + kk);
    v8bf a1 = *(const v8bf*)(arow + kk + 8);
    v16bf a = __builtin_shufflevector(a0, a1, 0,1,2,3,4,5,6,7,8,9,10,11,12,13,14,15);
    v16bf b = *(const v16bf*)(bcol + kk);
    acc = __builtin_amdgcn_wmma_f32_16x16x32_bf16(false, a, false, b, (short)0, acc, false, false);
  }
  int n = nt * 16 + lm;
  #pragma unroll
  for (int j = 0; j < 8; ++j) {
    int r = blockIdx.x * 16 + hi * 8 + j;
    out[(long long)r * NS + n] = (n < E_DIM) ? acc[j] : 0.f;
  }
}

// =====================================================================
// SpMM scatter: acc[row,:] += val * x[col,:]
__global__ void __launch_bounds__(256) k_spmm(
    const int* __restrict__ idx, const float* __restrict__ data,
    const float* __restrict__ x, float* __restrict__ acc, int nnz) {
  int sub = threadIdx.x >> 7, tid = threadIdx.x & 127;
  long long t = (long long)blockIdx.x * 2 + sub;
  if (t >= nnz) return;
  int row = idx[t], col = idx[(long long)nnz + t];
  float v = data[t];
  const float* xc = x + (long long)col * NS;
  float* ar = acc + (long long)row * NS;
  for (int d = tid; d < E_DIM; d += 128) atomicAdd(&ar[d], v * xc[d]);
}

// =====================================================================
// Highway: gate = sigmoid(glin + br[d]); out = gate*relu(y) + (1-gate)*ecur
__global__ void k_highway(const float* __restrict__ ecur, const float* __restrict__ y,
                          const float* __restrict__ glin, const float* __restrict__ br,
                          float* __restrict__ out, int ostride) {
  long long i = (long long)blockIdx.x * blockDim.x + threadIdx.x;
  if (i >= (long long)KG_E * NS) return;
  int r = (int)(i / NS), d = (int)(i % NS);
  if (d < E_DIM) {
    float g = 1.f / (1.f + expf(-(glin[i] + br[d])));
    float yy = y[i]; yy = yy > 0.f ? yy : 0.f;
    out[(long long)r * ostride + d] = g * yy + (1.f - g) * ecur[i];
  } else if (ostride == NS) {
    out[i] = 0.f;
  }
}

// =====================================================================
extern "C" void kernel_launch(void* const* d_in, const int* in_sizes, int n_in,
                              void* d_out, int out_size, void* d_ws, size_t ws_size,
                              hipStream_t stream) {
  const float* e0     = (const float*)d_in[0];
  const float* r_head = (const float*)d_in[1];
  const float* r_tail = (const float*)d_in[2];
  const float* eadjd  = (const float*)d_in[3];
  const float* beL    = (const float*)d_in[4];
  const float* beR    = (const float*)d_in[5];
  const float* atten  = (const float*)d_in[6];
  const float* gcnW1  = (const float*)d_in[7];
  const float* hwWr   = (const float*)d_in[8];
  const float* hwbr   = (const float*)d_in[9];
  const int*   eadji  = (const int*)d_in[10];
  const int*   eeri   = (const int*)d_in[11];
  const int*   eerr   = (const int*)d_in[12];
  float* out = (float*)d_out;

  // ---- workspace layout ----
  char* ws = (char*)d_ws;
  size_t o = 0;
  auto alloc = [&](size_t bytes) { o = (o + 255) & ~(size_t)255; size_t c = o; o += bytes; return c; };
  float*  rhinv  = (float*)(ws + alloc(512 * 4));
  float*  rtinv  = (float*)(ws + alloc(512 * 4));
  float*  rowsum = (float*)(ws + alloc((size_t)KG_E * 4));
  float*  aL     = (float*)(ws + alloc((size_t)512 * NS * 4));
  float*  aR     = (float*)(ws + alloc((size_t)512 * NS * 4));
  __bf16* WtG    = (__bf16*)(ws + alloc((size_t)NS * KAP * 2));
  __bf16* WtH    = (__bf16*)(ws + alloc((size_t)NS * KAP * 2));
  size_t btBytes = (size_t)NS * KRP * 2;                  // 30,409,728 each
  size_t btOff   = alloc(2 * btBytes);                    // BtL+BtR; later reused as Abf
  __bf16* BtL = (__bf16*)(ws + btOff);
  __bf16* BtR = (__bf16*)(ws + btOff + btBytes);
  __bf16* Abf = (__bf16*)(ws + btOff);                    // 32 MB <= 60.8 MB region
  size_t mBytes = (size_t)KG_E * NS * 4;                  // 60.8 MB
  float* accb = (float*)(ws + alloc(mBytes));
  float* bufA = (float*)(ws + alloc(mBytes));             // e1, later xW
  float* bufB = (float*)(ws + alloc(mBytes));             // e2, later g2 (in place)
  float* bufG = (float*)(ws + alloc(mBytes));             // gate linear
  (void)ws_size; (void)n_in; (void)in_sizes; (void)out_size;

  const int EW = (int)(((long long)KG_E * NS + 255) / 256);   // 59375
  dim3 gemRgrid(512 / 16, 3);                                 // 10 N-tile-pairs over 12 wave slots
  dim3 gem2grid(KG_E / 16, (NS / 16 + 3) / 4);                // (3125,5)
  dim3 btGrid((int)(KRP / 16), NS / 16);                      // (3126,19)

  // ---- constants precompute ----
  k_rowsum_inv<<<KG_R, 256, 0, stream>>>(r_head, rhinv, KG_E);
  k_rowsum_inv<<<KG_R, 256, 0, stream>>>(r_tail, rtinv, KG_E);
  k_cvt_wt<<<(NS * KAP + 255) / 256, 256, 0, stream>>>(gcnW1, WtG);
  k_cvt_wt<<<(NS * KAP + 255) / 256, 256, 0, stream>>>(hwWr, WtH);

  // ---- layer 1: r1 + attention (input e0, stride 300) ----
  k_make_bt<<<btGrid, dim3(16, 16), 0, stream>>>(e0, E_DIM, beL, beR, BtL, BtR);
  k_gemm_r<<<gemRgrid, 128, 0, stream>>>(r_head, BtL, rhinv, atten, aL);
  k_gemm_r<<<gemRgrid, 128, 0, stream>>>(r_tail, BtR, rtinv, atten + E_DIM, aR);
  hipMemsetAsync(accb, 0, mBytes, stream);
  hipMemsetAsync(rowsum, 0, (size_t)KG_E * 4, stream);
  k_att<<<NNZ / 8, 256, 0, stream>>>(e0, E_DIM, aL, aR, eeri, eerr, accb, rowsum, NNZ);
  k_att_combine<<<EW, 256, 0, stream>>>(e0, accb, rowsum, ALPHA1, bufA);   // e1

  // ---- layer 2: r2 + attention (input e1, stride NS) ----
  k_make_bt<<<btGrid, dim3(16, 16), 0, stream>>>(bufA, NS, beL, beR, BtL, BtR);
  k_gemm_r<<<gemRgrid, 128, 0, stream>>>(r_head, BtL, rhinv, atten, aL);
  k_gemm_r<<<gemRgrid, 128, 0, stream>>>(r_tail, BtR, rtinv, atten + E_DIM, aR);
  hipMemsetAsync(accb, 0, mBytes, stream);
  hipMemsetAsync(rowsum, 0, (size_t)KG_E * 4, stream);
  k_att<<<NNZ / 8, 256, 0, stream>>>(bufA, NS, aL, aR, eeri, eerr, accb, rowsum, NNZ);
  k_att_combine<<<EW, 256, 0, stream>>>(e0, accb, rowsum, ALPHA2, bufB);   // e2

  // ---- GCN stage 1: g1 = relu(spmm(e2 @ W)); g2 = highway(e2, g1) in place ----
  k_cvt_a<<<(int)(((long long)KG_E * KAP + 255) / 256), 256, 0, stream>>>(bufB, Abf);
  k_gemm2<<<gem2grid, 128, 0, stream>>>(Abf, WtG, bufA);                   // xW
  k_gemm2<<<gem2grid, 128, 0, stream>>>(Abf, WtH, bufG);                   // gate lin
  hipMemsetAsync(accb, 0, mBytes, stream);
  k_spmm<<<NNZ / 2, 256, 0, stream>>>(eadji, eadjd, bufA, accb, NNZ);
  k_highway<<<EW, 256, 0, stream>>>(bufB, accb, bufG, hwbr, bufB, NS);     // g2 in place

  // ---- GCN stage 2: g3 = relu(spmm(g2 @ W)); out = highway(g2, g3) ----
  k_cvt_a<<<(int)(((long long)KG_E * KAP + 255) / 256), 256, 0, stream>>>(bufB, Abf);
  k_gemm2<<<gem2grid, 128, 0, stream>>>(Abf, WtG, bufA);
  k_gemm2<<<gem2grid, 128, 0, stream>>>(Abf, WtH, bufG);
  hipMemsetAsync(accb, 0, mBytes, stream);
  k_spmm<<<NNZ / 2, 256, 0, stream>>>(eadji, eadjd, bufA, accb, NNZ);
  k_highway<<<EW, 256, 0, stream>>>(bufB, accb, bufG, hwbr, out, E_DIM);
}
